// GRUNet_11510512353655
// MI455X (gfx1250) — compile-verified, matrix operand order checked by reading, not run on hardware
//
#include <hip/hip_runtime.h>

#define HIDDEN  64
#define B_TOT   256
#define T_LEN   3000
#define TCHUNK  100      // 3000 = 30 * 100
#define ROWS    16       // batch rows per block (= WMMA M)
#define BLOCK   128      // 4 waves; each wave owns 16 hidden columns

typedef __attribute__((ext_vector_type(16))) _Float16 v16h;
typedef __attribute__((ext_vector_type(8)))  _Float16 v8h;
typedef __attribute__((ext_vector_type(8)))  float    v8f;

// Native gfx1250 V_TANH_F32 if the toolchain exposes it; otherwise fast
// exp/rcp (single TRANS32 ops, no IEEE division fixup chain).
#if __has_builtin(__builtin_amdgcn_tanhf)
__device__ __forceinline__ float tanh_f(float x) {
    return __builtin_amdgcn_tanhf(x);
}
#elif __has_builtin(__builtin_amdgcn_tanh_f32)
__device__ __forceinline__ float tanh_f(float x) {
    return __builtin_amdgcn_tanh_f32(x);
}
#else
__device__ __forceinline__ float tanh_f(float x) {
    return 1.0f - 2.0f * __builtin_amdgcn_rcpf(1.0f + __expf(2.0f * x));
}
#endif

#if __has_builtin(__builtin_amdgcn_tanhf) || __has_builtin(__builtin_amdgcn_tanh_f32)
// sigmoid(x) = 0.5*tanh(x/2) + 0.5 : one TRANS op instead of exp+add+rcp
__device__ __forceinline__ float sigmoid_f(float x) {
    return __builtin_fmaf(0.5f, tanh_f(0.5f * x), 0.5f);
}
#else
__device__ __forceinline__ float sigmoid_f(float x) {
    return __builtin_amdgcn_rcpf(1.0f + __expf(-x));
}
#endif

__global__ __launch_bounds__(BLOCK) void gru_wmma_kernel(
    const float* __restrict__ x,     // (B, T, 1)
    const float* __restrict__ W_ih,  // (3H, 1)
    const float* __restrict__ W_hh,  // (3H, H)
    const float* __restrict__ b_ih,  // (3H,)
    const float* __restrict__ b_hh,  // (3H,)
    const float* __restrict__ W_fc,  // (1, H)
    const float* __restrict__ b_fc,  // (1,)
    float* __restrict__ out)         // (B,)
{
    // h exchange buffers, stored directly in WMMA A-fragment layout:
    //   4 K-tiles of 16x16 f16; tile k: lane l holds 8 halfs (16B) = one ds_load_b128.
    __shared__ v8h   hbuf[2][4][32];
    __shared__ float xbuf[ROWS * TCHUNK];
    __shared__ float outacc[ROWS];

    const int tid     = threadIdx.x;
    const int wave    = tid >> 5;
    const int lane    = tid & 31;
    const int nidx    = lane & 15;          // N within 16-wide tile
    const int m0      = (lane >> 4) << 3;   // C/D row base: 0 or 8
    const int colbase = wave * 16;          // this wave's hidden-column tile
    const int col     = colbase + nidx;
    const int rowbase = blockIdx.x * ROWS;

    // ---- per-lane scalar constants -----------------------------------------
    const float wir  = W_ih[col];
    const float wiz  = W_ih[64 + col];
    const float win  = W_ih[128 + col];
    const float brc  = b_ih[col]      + b_hh[col];        // folded r bias
    const float bzc  = b_ih[64 + col] + b_hh[64 + col];   // folded z bias
    const float binc = b_ih[128 + col];                   // input n bias
    const float bhn  = b_hh[128 + col];                   // hidden n bias (inside r*(...))
    const float wfc  = W_fc[col];

    // ---- preload W_hh^T as WMMA B fragments (constant over all T steps) ----
    // B[k][n] = W_hh[g*64 + col][k_abs]; layout mirrors A: lanes 0-15 halfs
    // 0..7 -> K 0..7, halfs 8..15 -> K 16..23; lanes 16-31 shifted by +8.
    v16h Bf[3][2];
#pragma unroll
    for (int g = 0; g < 3; ++g) {
#pragma unroll
        for (int c = 0; c < 2; ++c) {
            v16h bb;
#pragma unroll
            for (int hh = 0; hh < 16; ++hh) {
                int kr = (hh < 8) ? hh : (hh + 8);
                if (lane >= 16) kr += 8;
                bb[hh] = (_Float16)W_hh[(g * 64 + col) * HIDDEN + c * 32 + kr];
            }
            Bf[g][c] = bb;
        }
    }

    if (tid < ROWS) outacc[tid] = 0.0f;

    // ---- state: h in C/D fragment layout, fp32: h[v] = h(row m0+v, col) ----
    v8f h = {};

    const int hiK   = (lane & 8) ? 16 : 0;  // K_rel>=8 lives in lanes 16..31
    const int halfi = lane & 7;

    for (int t = 0; t < T_LEN; ++t) {
        const int tc = t % TCHUNK;
        if (tc == 0) {
            __syncthreads();  // protect xbuf readers from previous chunk
            for (int i = tid; i < ROWS * TCHUNK; i += BLOCK) {
                const int m  = i / TCHUNK;
                const int tt = i - m * TCHUNK;
                const size_t gidx = (size_t)(rowbase + m) * T_LEN + (t + tt);
                xbuf[i] = x[gidx];
                if (t + TCHUNK < T_LEN)
                    __builtin_prefetch(&x[gidx + TCHUNK], 0, 1);  // global_prefetch_b8
            }
        }

        // ---- scatter h (f16) into A-fragment layout (own K-tile == wave) ---
        const int p = t & 1;
        {
            _Float16* hb = (_Float16*)&hbuf[p][wave][0];
#pragma unroll
            for (int v = 0; v < 8; ++v) {
                const int tl = m0 + v + hiK;          // target lane for (m, K_rel)
                hb[tl * 8 + halfi] = (_Float16)h[v];  // ds_store_b16, stride 16B
            }
        }
        __syncthreads();

        // ---- gather full A (16x64) as two v16h fragments: 4x ds_load_b128 --
        const v8h a0 = hbuf[p][0][lane];
        const v8h a1 = hbuf[p][1][lane];
        const v8h a2 = hbuf[p][2][lane];
        const v8h a3 = hbuf[p][3][lane];
        const v16h A0 = __builtin_shufflevector(a0, a1, 0,1,2,3,4,5,6,7,8,9,10,11,12,13,14,15);
        const v16h A1 = __builtin_shufflevector(a2, a3, 0,1,2,3,4,5,6,7,8,9,10,11,12,13,14,15);

        // ---- fold input-gate terms + biases into the WMMA C operands -------
        // accR/accZ seed = x*W_ih + (b_ih+b_hh); accN seed = b_hh[n] (scaled by
        // r AFTER the matmul per torch GRU semantics). i_n kept separately.
        v8f accR, accZ, accN;
        float i_n[8];
        float xv[8];
#pragma unroll
        for (int v = 0; v < 8; ++v) {
            xv[v]   = xbuf[(m0 + v) * TCHUNK + tc];
            accR[v] = __builtin_fmaf(xv[v], wir, brc);
            accZ[v] = __builtin_fmaf(xv[v], wiz, bzc);
            accN[v] = bhn;
            i_n[v]  = __builtin_fmaf(xv[v], win, binc);
        }

        // ---- gh = h @ W_hh^T : three gate tiles, K=64 = 2 WMMAs each -------
        accR = __builtin_amdgcn_wmma_f32_16x16x32_f16(false, A0, false, Bf[0][0], (short)0, accR, false, false);
        accR = __builtin_amdgcn_wmma_f32_16x16x32_f16(false, A1, false, Bf[0][1], (short)0, accR, false, false);
        accZ = __builtin_amdgcn_wmma_f32_16x16x32_f16(false, A0, false, Bf[1][0], (short)0, accZ, false, false);
        accZ = __builtin_amdgcn_wmma_f32_16x16x32_f16(false, A1, false, Bf[1][1], (short)0, accZ, false, false);
        accN = __builtin_amdgcn_wmma_f32_16x16x32_f16(false, A0, false, Bf[2][0], (short)0, accN, false, false);
        accN = __builtin_amdgcn_wmma_f32_16x16x32_f16(false, A1, false, Bf[2][1], (short)0, accN, false, false);

        // ---- gates + state update (torch order r, z, n) --------------------
#pragma unroll
        for (int v = 0; v < 8; ++v) {
            const float r = sigmoid_f(accR[v]);
            const float z = sigmoid_f(accZ[v]);
            const float n = tanh_f(__builtin_fmaf(r, accN[v], i_n[v]));
            h[v] = __builtin_fmaf(z, h[v] - n, n);   // (1-z)*n + z*h
        }
    }

    // ---- out[b] = hT[b,:] . W_fc + b_fc : LDS float-atomic reduction -------
#pragma unroll
    for (int v = 0; v < 8; ++v)
        atomicAdd(&outacc[m0 + v], h[v] * wfc);
    __syncthreads();
    if (tid < ROWS)
        out[rowbase + tid] = outacc[tid] + b_fc[0];
}

extern "C" void kernel_launch(void* const* d_in, const int* in_sizes, int n_in,
                              void* d_out, int out_size, void* d_ws, size_t ws_size,
                              hipStream_t stream) {
    const float* x    = (const float*)d_in[0];
    const float* W_ih = (const float*)d_in[1];
    const float* W_hh = (const float*)d_in[2];
    const float* b_ih = (const float*)d_in[3];
    const float* b_hh = (const float*)d_in[4];
    const float* W_fc = (const float*)d_in[5];
    const float* b_fc = (const float*)d_in[6];
    float* out = (float*)d_out;

    dim3 grid(B_TOT / ROWS);   // 16 blocks, one per 16-row batch slice
    dim3 block(BLOCK);         // 4 wave32s
    hipLaunchKernelGGL(gru_wmma_kernel, grid, block, 0, stream,
                       x, W_ih, W_hh, b_ih, b_hh, W_fc, b_fc, out);
}
